// TransformerBlock_10204842295831
// MI455X (gfx1250) — matrix operand order repaired and checked by reading, not compiled, not run on hardware
//
#include <hip/hip_runtime.h>
#include <hip/hip_bf16.h>
#include <math.h>

typedef __attribute__((ext_vector_type(16))) _Float16 v16h;
typedef __attribute__((ext_vector_type(8)))  _Float16 h8;
typedef __attribute__((ext_vector_type(8)))  float    v8f;

#define N_PIX 4096
#define HGT 64
#define WID 64

// ---------------------------------------------------------------- WMMA core

__device__ __forceinline__ v8f wmma_f16(v16h a, v16h b, v8f c) {
  // D = A(16x32 f16) * B(32x16 f16) + C(16x16 f32)
  return __builtin_amdgcn_wmma_f32_16x16x32_f16(false, a, false, b, (short)0, c,
                                                false, false);
}

// A fragment: lane l: row m = l&15, group g = l>>4.
// half h<8  -> K = k0 + 16g + h       (contiguous 8 halves)
// half h>=8 -> K = k0 + 16g + h       (contiguous 8 halves)
__device__ __forceinline__ v16h load_a_frag(const _Float16* A, int row, int lda,
                                            int k0, int g) {
  const h8 lo = *(const h8*)(A + (size_t)row * lda + k0 + 16 * g);
  const h8 hi = *(const h8*)(A + (size_t)row * lda + k0 + 16 * g + 8);
  v16h r;
#pragma unroll
  for (int i = 0; i < 8; ++i) { r[i] = lo[i]; r[i + 8] = hi[i]; }
  return r;
}

// B operand pre-packed fragment-major: dst[(((nt*(K/16)+c)*16)+n)*16+h]
// holds B[k = c*16+h][col = nt*16+n].  Lane (n=nl, group g) needs halves
// h=0..15 with K = k0+16g+h  ->  chunk c = k0/16+g, 32 contiguous bytes.
__device__ __forceinline__ v16h load_b_pk(const _Float16* Bpk, int K, int nt,
                                          int k0, int nl, int g) {
  int chunk = (k0 >> 4) + g;
  const _Float16* p =
      Bpk + ((size_t)(nt * (K >> 4) + chunk) * 16 + nl) * 16;
  const h8 lo = *(const h8*)p;
  const h8 hi = *(const h8*)(p + 8);
  v16h r;
#pragma unroll
  for (int i = 0; i < 8; ++i) { r[i] = lo[i]; r[i + 8] = hi[i]; }
  return r;
}

// C[pixels x Ncols] = A[pixels x K (f16, row-major)] * Bpk (fragment-major)
__device__ __forceinline__ v8f gemm_core(const _Float16* A, const _Float16* Bpk,
                                         int K, int arow, int nt, int nl, int g) {
  __builtin_prefetch(A + (size_t)arow * K, 0, 3);
  __builtin_prefetch(Bpk + (size_t)nt * K * 16, 0, 3);
  v8f acc = {};
#pragma unroll
  for (int k0 = 0; k0 < K; k0 += 32) {
    v16h a = load_a_frag(A, arow, K, k0, g);
    v16h b = load_b_pk(Bpk, K, nt, k0, nl, g);
    acc = wmma_f16(a, b, acc);
  }
  return acc;
}

// ---------------------------------------------------------------- weight pack

// Pack weight W[O][K] (1x1 conv OIHW) into fragment-major B tiles for
// C = A[M x K] * B[K x O].
__device__ __forceinline__ void pack_b(const float* W, _Float16* dst, int K,
                                       int N, int tid, int nth) {
  int chunks = K >> 4;
  int total = K * N;
  for (int i = tid; i < total; i += nth) {
    int h = i & 15;
    int n = (i >> 4) & 15;
    int rest = i >> 8;
    int c = rest % chunks;
    int nt = rest / chunks;
    int k = c * 16 + h;
    int col = nt * 16 + n;
    dst[i] = (_Float16)W[(size_t)col * K + k];
  }
}

__global__ void pack_kernel(const float* qkv_w, const float* proj_w,
                            const float* gam_w, const float* gam_b,
                            const float* bet_w, const float* bet_b,
                            const float* ffn1_w, const float* ffn2_w,
                            _Float16* wqkv, _Float16* wproj, _Float16* wgb,
                            float* gbb, _Float16* wffn1, _Float16* wffn2) {
  int tid = blockIdx.x * blockDim.x + threadIdx.x;
  int nth = gridDim.x * blockDim.x;
  pack_b(qkv_w, wqkv, 64, 192, tid, nth);
  pack_b(proj_w, wproj, 64, 64, tid, nth);
  pack_b(ffn1_w, wffn1, 64, 256, tid, nth);
  pack_b(ffn2_w, wffn2, 256, 64, tid, nth);
  // gamma|beta combined: K=32, N=128 (cols 0-63 gamma, 64-127 beta)
  for (int i = tid; i < 32 * 128; i += nth) {
    int h = i & 15;
    int n = (i >> 4) & 15;
    int rest = i >> 8;
    int c = rest % 2;          // chunks = 32/16 = 2
    int nt = rest / 2;
    int k = c * 16 + h;
    int col = nt * 16 + n;
    float v = (col < 64) ? gam_w[col * 32 + k] : bet_w[(col - 64) * 32 + k];
    wgb[i] = (_Float16)v;
  }
  for (int i = tid; i < 128; i += nth) gbb[i] = (i < 64) ? gam_b[i] : bet_b[i - 64];
}

// ---------------------------------------------------------------- LayerNorm

__global__ __launch_bounds__(256) void ln_kernel(const float* __restrict__ src,
                                                 const float* w, const float* b,
                                                 _Float16* dst) {
  int p = blockIdx.x * 256 + threadIdx.x;   // pixel; src is [64][4096]
  float v[64];
  float mu = 0.f;
#pragma unroll
  for (int c = 0; c < 64; ++c) { v[c] = src[(size_t)c * N_PIX + p]; mu += v[c]; }
  mu *= (1.f / 64.f);
  float var = 0.f;
#pragma unroll
  for (int c = 0; c < 64; ++c) { float d = v[c] - mu; var += d * d; }
  var *= (1.f / 64.f);
  float rs = rsqrtf(var + 1e-5f);
#pragma unroll
  for (int c = 0; c < 64; ++c)
    dst[(size_t)p * 64 + c] = (_Float16)((v[c] - mu) * rs * w[c] + b[c]);
}

// ---------------------------------------------------------------- luma path

__global__ __launch_bounds__(256) void lconv1_kernel(const float* luma,
                                                     const float* w, const float* b,
                                                     float* h1) {
  int idx = blockIdx.x * 256 + threadIdx.x;   // 32*4096
  int ch = idx >> 12, p = idx & 4095;
  int y = p >> 6, x = p & 63;
  float s = b[ch];
#pragma unroll
  for (int dy = 0; dy < 3; ++dy)
#pragma unroll
    for (int dx = 0; dx < 3; ++dx) {
      int yy = y + dy - 1, xx = x + dx - 1;
      if (yy >= 0 && yy < HGT && xx >= 0 && xx < WID)
        s += luma[yy * WID + xx] * w[ch * 9 + dy * 3 + dx];
    }
  h1[(size_t)ch * N_PIX + p] = fmaxf(s, 0.f);
}

__global__ __launch_bounds__(256) void lconv2_kernel(const float* h1,
                                                     const float* w, const float* b,
                                                     _Float16* h2h) {
  int idx = blockIdx.x * 256 + threadIdx.x;   // 32*4096
  int co = idx >> 12, p = idx & 4095;
  int y = p >> 6, x = p & 63;
  float s = b[co];
  for (int ci = 0; ci < 32; ++ci) {
    const float* hp = h1 + (size_t)ci * N_PIX;
    const float* wp = w + co * 288 + ci * 9;
#pragma unroll
    for (int dy = 0; dy < 3; ++dy)
#pragma unroll
      for (int dx = 0; dx < 3; ++dx) {
        int yy = y + dy - 1, xx = x + dx - 1;
        if (yy >= 0 && yy < HGT && xx >= 0 && xx < WID)
          s += hp[yy * WID + xx] * wp[dy * 3 + dx];
      }
  }
  h2h[(size_t)p * 32 + co] = (_Float16)fmaxf(s, 0.f);   // [N][32] for WMMA A
}

// avgpool3x3(1-luma) (zero pad, /9) + deterministic partial sums
__global__ __launch_bounds__(256) void pool_kernel(const float* luma,
                                                   float* invLp, float* part) {
  __shared__ float red[256];
  int p = blockIdx.x * 256 + threadIdx.x;
  int y = p >> 6, x = p & 63;
  float s = 0.f;
#pragma unroll
  for (int dy = 0; dy < 3; ++dy)
#pragma unroll
    for (int dx = 0; dx < 3; ++dx) {
      int yy = y + dy - 1, xx = x + dx - 1;
      if (yy >= 0 && yy < HGT && xx >= 0 && xx < WID)
        s += 1.f - luma[yy * WID + xx];
    }
  s *= (1.f / 9.f);
  invLp[p] = s;
  red[threadIdx.x] = s;
  __syncthreads();
  for (int off = 128; off > 0; off >>= 1) {
    if (threadIdx.x < off) red[threadIdx.x] += red[threadIdx.x + off];
    __syncthreads();
  }
  if (threadIdx.x == 0) part[blockIdx.x] = red[0];
}

__global__ void pool_reduce_kernel(const float* part, float* invsum) {
  float s = 0.f;
  for (int i = 0; i < 16; ++i) s += part[i];   // fixed order -> deterministic
  invsum[0] = s;
}

// ---------------------------------------------------------------- GEMMs

// gamma/beta: [4096x32] x [32x128] -> g_buf/bt_buf [64][4096] f32
__global__ __launch_bounds__(256) void gemm_gb_kernel(const _Float16* A,
                                                      const _Float16* Bpk,
                                                      const float* gbb,
                                                      float* gbuf, float* btbuf) {
  int lane = threadIdx.x & 31, w = threadIdx.x >> 5;
  int tile = blockIdx.x * 8 + w;
  int mt = tile / 8, nt = tile % 8;
  int nl = lane & 15, g = lane >> 4;
  v8f acc = gemm_core(A, Bpk, 32, mt * 16 + nl, nt, nl, g);
  int col = nt * 16 + nl;
  float bias = gbb[col];
  float* dst = (col < 64) ? (gbuf + (size_t)col * N_PIX)
                          : (btbuf + (size_t)(col - 64) * N_PIX);
#pragma unroll
  for (int j = 0; j < 8; ++j) dst[mt * 16 + j + 8 * g] = acc[j] + bias;
}

// qkv: [4096x64] x [64x192]; epilogue: bias, g*x+bt conditioning,
// q += alpha*invLc, q *= dh^-0.5; q,k -> [head][N][8] f16; v -> [head][8][N]
__global__ __launch_bounds__(256) void gemm_qkv_kernel(
    const _Float16* A, const _Float16* Bpk, const float* qkvb, const float* gbuf,
    const float* btbuf, const float* invLp, const float* invsum,
    const float* alphap, _Float16* qh, _Float16* kh, _Float16* vT) {
  int lane = threadIdx.x & 31, w = threadIdx.x >> 5;
  int tile = blockIdx.x * 8 + w;
  int mt = tile / 12, nt = tile % 12;
  int nl = lane & 15, g = lane >> 4;
  v8f acc = gemm_core(A, Bpk, 64, mt * 16 + nl, nt, nl, g);
  int col = nt * 16 + nl;
  int which = col >> 6, ch = col & 63, head = ch >> 3, d = ch & 7;
  float bias = qkvb[col];
  float mean = invsum[0] * (1.f / 4096.f);
  float al = alphap[0];
#pragma unroll
  for (int j = 0; j < 8; ++j) {
    int pix = mt * 16 + j + 8 * g;
    float val = acc[j] + bias;
    val = gbuf[(size_t)ch * N_PIX + pix] * val + btbuf[(size_t)ch * N_PIX + pix];
    if (which == 0) {
      val = (val + al * (invLp[pix] - mean)) * 0.35355339059f;  // fold 1/sqrt(8)
      qh[((size_t)head * N_PIX + pix) * 8 + d] = (_Float16)val;
    } else if (which == 1) {
      kh[((size_t)head * N_PIX + pix) * 8 + d] = (_Float16)val;
    } else {
      vT[((size_t)head * 8 + d) * N_PIX + pix] = (_Float16)val;
    }
  }
}

// proj: [4096x64] x [64x64]; + bias + residual x -> x2 [64][4096]
__global__ __launch_bounds__(256) void gemm_proj_kernel(const _Float16* A,
                                                        const _Float16* Bpk,
                                                        const float* pb,
                                                        const float* x, float* x2) {
  int lane = threadIdx.x & 31, w = threadIdx.x >> 5;
  int tile = blockIdx.x * 8 + w;
  int mt = tile / 4, nt = tile % 4;
  int nl = lane & 15, g = lane >> 4;
  v8f acc = gemm_core(A, Bpk, 64, mt * 16 + nl, nt, nl, g);
  int c = nt * 16 + nl;
  float bias = pb[c];
#pragma unroll
  for (int j = 0; j < 8; ++j) {
    int pix = mt * 16 + j + 8 * g;
    x2[(size_t)c * N_PIX + pix] = x[(size_t)c * N_PIX + pix] + acc[j] + bias;
  }
}

// ffn1: [4096x64] x [64x256] + bias -> y1 [256][4096] (channel major for dwconv)
__global__ __launch_bounds__(256) void gemm_ffn1_kernel(const _Float16* A,
                                                        const _Float16* Bpk,
                                                        const float* fb, float* y1) {
  int lane = threadIdx.x & 31, w = threadIdx.x >> 5;
  int tile = blockIdx.x * 8 + w;
  int mt = tile / 16, nt = tile % 16;
  int nl = lane & 15, g = lane >> 4;
  v8f acc = gemm_core(A, Bpk, 64, mt * 16 + nl, nt, nl, g);
  int c = nt * 16 + nl;
  float bias = fb[c];
#pragma unroll
  for (int j = 0; j < 8; ++j)
    y1[(size_t)c * N_PIX + mt * 16 + j + 8 * g] = acc[j] + bias;
}

// ffn2: [4096x256] x [256x64] + bias + residual x2 -> d_out [64][4096]
__global__ __launch_bounds__(256) void gemm_ffn2_kernel(const _Float16* A,
                                                        const _Float16* Bpk,
                                                        const float* fb,
                                                        const float* x2,
                                                        float* out) {
  int lane = threadIdx.x & 31, w = threadIdx.x >> 5;
  int tile = blockIdx.x * 8 + w;
  int mt = tile / 4, nt = tile % 4;
  int nl = lane & 15, g = lane >> 4;
  v8f acc = gemm_core(A, Bpk, 256, mt * 16 + nl, nt, nl, g);
  int c = nt * 16 + nl;
  float bias = fb[c];
#pragma unroll
  for (int j = 0; j < 8; ++j) {
    int pix = mt * 16 + j + 8 * g;
    out[(size_t)c * N_PIX + pix] = x2[(size_t)c * N_PIX + pix] + acc[j] + bias;
  }
}

// ---------------------------------------------------------------- attention
// Flash attention: 1 wave per (head, 16-query block), 32 keys per step.
// q/k: [head][N][8] f16 (dh padded to K=32 with zeros); v: [head][8][N] f16.
__global__ __launch_bounds__(256) void attn_kernel(const _Float16* __restrict__ qh,
                                                   const _Float16* __restrict__ kh,
                                                   const _Float16* __restrict__ vT,
                                                   _Float16* attn_h) {
  __shared__ float    sS[8][16][32];
  __shared__ _Float16 sP[8][16][32];
  __shared__ float    sM[8][16];
  __shared__ float    sL[8][16];
  __shared__ float    sC[8][16];

  int lane = threadIdx.x & 31, w = threadIdx.x >> 5;
  int wg = blockIdx.x * 8 + w;        // 2048 waves
  int head = wg >> 8;                 // 256 query blocks per head
  int qb = wg & 255;
  int nl = lane & 15, g = lane >> 4;

  // Q fragment: rows = queries, K halves 0..7 = dh (rest zero padding)
  v16h qa = {};
  if (g == 0) {
    const h8 qr = *(const h8*)(qh + ((size_t)head * N_PIX + qb * 16 + nl) * 8);
#pragma unroll
    for (int i = 0; i < 8; ++i) qa[i] = qr[i];
  }
  if (lane < 16) { sM[w][lane] = -3.0e38f; sL[w][lane] = 0.f; }

  v8f o = {};
  for (int kb = 0; kb < 128; ++kb) {
    int keyb = kb * 32;
    // scores for two 16-key sub-tiles
    v16h b0 = {}, b1 = {};
    if (g == 0) {
      const h8 k0 = *(const h8*)(kh + ((size_t)head * N_PIX + keyb + nl) * 8);
      const h8 k1 = *(const h8*)(kh + ((size_t)head * N_PIX + keyb + 16 + nl) * 8);
#pragma unroll
      for (int i = 0; i < 8; ++i) { b0[i] = k0[i]; b1[i] = k1[i]; }
    }
    v8f s0 = {}, s1 = {};
    s0 = wmma_f16(qa, b0, s0);
    s1 = wmma_f16(qa, b1, s1);
#pragma unroll
    for (int j = 0; j < 8; ++j) {
      sS[w][j + 8 * g][nl] = s0[j];
      sS[w][j + 8 * g][16 + nl] = s1[j];
    }
    asm volatile("" ::: "memory");   // intra-wave LDS ordering (DS in-order)
    if (lane < 16) {                 // per-row online softmax
      float mold = sM[w][lane];
      float cm = mold;
#pragma unroll
      for (int jj = 0; jj < 32; ++jj) cm = fmaxf(cm, sS[w][lane][jj]);
      float corr = __expf(mold - cm);
      float l = sL[w][lane] * corr;
#pragma unroll
      for (int jj = 0; jj < 32; ++jj) {
        float pe = __expf(sS[w][lane][jj] - cm);
        sP[w][lane][jj] = (_Float16)pe;
        l += pe;
      }
      sM[w][lane] = cm; sL[w][lane] = l; sC[w][lane] = corr;
    }
    asm volatile("" ::: "memory");
    // rescale accumulator rows
#pragma unroll
    for (int j = 0; j < 8; ++j) o[j] *= sC[w][j + 8 * g];
    // P fragment (A layout): lane row = query, halves map key 0..31
    v16h pa;
#pragma unroll
    for (int h = 0; h < 8; ++h) {
      pa[h] = sP[w][nl][16 * g + h];
      pa[8 + h] = sP[w][nl][16 * g + 8 + h];
    }
    // V fragment (B layout): col n = dh (cols 8..15 zero), K = key
    v16h bv = {};
    if (nl < 8) {
      const _Float16* vp = vT + ((size_t)head * 8 + nl) * N_PIX + keyb + 16 * g;
      const h8 v0 = *(const h8*)(vp);
      const h8 v1 = *(const h8*)(vp + 8);
#pragma unroll
      for (int i = 0; i < 8; ++i) { bv[i] = v0[i]; bv[8 + i] = v1[i]; }
    }
    o = wmma_f16(pa, bv, o);
  }
  asm volatile("" ::: "memory");
  if (nl < 8) {                       // normalize, write [N][64] f16 for proj
#pragma unroll
    for (int j = 0; j < 8; ++j) {
      int pix = qb * 16 + j + 8 * g;
      float val = o[j] / sL[w][j + 8 * g];
      attn_h[(size_t)pix * 64 + head * 8 + nl] = (_Float16)val;
    }
  }
}

// ---------------------------------------------------------------- dwconv+GELU

__global__ __launch_bounds__(256) void dwconv_gelu_kernel(const float* y1,
                                                          const float* w,
                                                          const float* b,
                                                          _Float16* y2) {
  int idx = blockIdx.x * 256 + threadIdx.x;   // 256*4096
  int ch = idx >> 12, p = idx & 4095;
  int y = p >> 6, x = p & 63;
  float s = b[ch];
  const float* yp = y1 + (size_t)ch * N_PIX;
  const float* wp = w + ch * 9;
#pragma unroll
  for (int dy = 0; dy < 3; ++dy)
#pragma unroll
    for (int dx = 0; dx < 3; ++dx) {
      int yy = y + dy - 1, xx = x + dx - 1;
      if (yy >= 0 && yy < HGT && xx >= 0 && xx < WID)
        s += yp[yy * WID + xx] * wp[dy * 3 + dx];
    }
  float ge = 0.5f * s * (1.f + erff(s * 0.70710678f));   // exact GELU
  y2[(size_t)p * 256 + ch] = (_Float16)ge;               // [N][256] for ffn2 A
}

// ---------------------------------------------------------------- launch

extern "C" void kernel_launch(void* const* d_in, const int* in_sizes, int n_in,
                              void* d_out, int out_size, void* d_ws, size_t ws_size,
                              hipStream_t stream) {
  const float* x      = (const float*)d_in[0];
  const float* luma   = (const float*)d_in[1];
  const float* ln1_w  = (const float*)d_in[2];
  const float* ln1_b  = (const float*)d_in[3];
  const float* qkv_w  = (const float*)d_in[4];
  const float* qkv_b  = (const float*)d_in[5];
  const float* proj_w = (const float*)d_in[6];
  const float* proj_b = (const float*)d_in[7];
  const float* lc1_w  = (const float*)d_in[8];
  const float* lc1_b  = (const float*)d_in[9];
  const float* lc2_w  = (const float*)d_in[10];
  const float* lc2_b  = (const float*)d_in[11];
  const float* gam_w  = (const float*)d_in[12];
  const float* gam_b  = (const float*)d_in[13];
  const float* bet_w  = (const float*)d_in[14];
  const float* bet_b  = (const float*)d_in[15];
  const float* alpha  = (const float*)d_in[16];
  const float* ln2_w  = (const float*)d_in[17];
  const float* ln2_b  = (const float*)d_in[18];
  const float* ffn1_w = (const float*)d_in[19];
  const float* ffn1_b = (const float*)d_in[20];
  const float* dw_w   = (const float*)d_in[21];
  const float* dw_b   = (const float*)d_in[22];
  const float* ffn2_w = (const float*)d_in[23];
  const float* ffn2_b = (const float*)d_in[24];
  float* out = (float*)d_out;

  char* base = (char*)d_ws;
  size_t off = 0;
  auto alloc = [&](size_t bytes) -> void* {
    void* p = base + off;
    off = (off + bytes + 255) & ~(size_t)255;
    return p;
  };
  _Float16* wqkv  = (_Float16*)alloc(64 * 192 * 2);
  _Float16* wproj = (_Float16*)alloc(64 * 64 * 2);
  _Float16* wgb   = (_Float16*)alloc(32 * 128 * 2);
  _Float16* wffn1 = (_Float16*)alloc(64 * 256 * 2);
  _Float16* wffn2 = (_Float16*)alloc(256 * 64 * 2);
  float*    gbb   = (float*)alloc(128 * 4);
  _Float16* xn_h  = (_Float16*)alloc((size_t)N_PIX * 64 * 2);
  float*    h1    = (float*)alloc((size_t)32 * N_PIX * 4);
  _Float16* h2_h  = (_Float16*)alloc((size_t)N_PIX * 32 * 2);
  float*    invLp = (float*)alloc(N_PIX * 4);
  float*    part  = (float*)alloc(16 * 4);
  float*    invsum= (float*)alloc(4);
  float*    gbuf  = (float*)alloc((size_t)64 * N_PIX * 4);
  float*    btbuf = (float*)alloc((size_t)64 * N_PIX * 4);
  _Float16* q_h   = (_Float16*)alloc((size_t)8 * N_PIX * 8 * 2);
  _Float16* k_h   = (_Float16*)alloc((size_t)8 * N_PIX * 8 * 2);
  _Float16* vT_h  = (_Float16*)alloc((size_t)8 * 8 * N_PIX * 2);
  _Float16* attn_h= (_Float16*)alloc((size_t)N_PIX * 64 * 2);
  float*    x2    = (float*)alloc((size_t)64 * N_PIX * 4);
  _Float16* y_h   = (_Float16*)alloc((size_t)N_PIX * 64 * 2);
  float*    y1    = (float*)alloc((size_t)256 * N_PIX * 4);
  _Float16* y2_h  = (_Float16*)alloc((size_t)N_PIX * 256 * 2);
  (void)ws_size; (void)n_in; (void)in_sizes; (void)out_size;

  pack_kernel<<<64, 256, 0, stream>>>(qkv_w, proj_w, gam_w, gam_b, bet_w, bet_b,
                                      ffn1_w, ffn2_w, wqkv, wproj, wgb, gbb,
                                      wffn1, wffn2);
  ln_kernel<<<16, 256, 0, stream>>>(x, ln1_w, ln1_b, xn_h);
  lconv1_kernel<<<512, 256, 0, stream>>>(luma, lc1_w, lc1_b, h1);
  lconv2_kernel<<<512, 256, 0, stream>>>(h1, lc2_w, lc2_b, h2_h);
  pool_kernel<<<16, 256, 0, stream>>>(luma, invLp, part);
  pool_reduce_kernel<<<1, 1, 0, stream>>>(part, invsum);
  gemm_gb_kernel<<<256, 256, 0, stream>>>(h2_h, wgb, gbb, gbuf, btbuf);
  gemm_qkv_kernel<<<384, 256, 0, stream>>>(xn_h, wqkv, qkv_b, gbuf, btbuf,
                                           invLp, invsum, alpha, q_h, k_h, vT_h);
  attn_kernel<<<256, 256, 0, stream>>>(q_h, k_h, vT_h, attn_h);
  gemm_proj_kernel<<<128, 256, 0, stream>>>(attn_h, wproj, proj_b, x, x2);
  ln_kernel<<<16, 256, 0, stream>>>(x2, ln2_w, ln2_b, y_h);
  gemm_ffn1_kernel<<<512, 256, 0, stream>>>(y_h, wffn1, ffn1_b, y1);
  dwconv_gelu_kernel<<<4096, 256, 0, stream>>>(y1, dw_w, dw_b, y2_h);
  gemm_ffn2_kernel<<<128, 256, 0, stream>>>(y2_h, wffn2, ffn2_b, x2, out);
}
